// TradeNet_52450140618954
// MI455X (gfx1250) — matrix operand order repaired and checked by reading, not compiled, not run on hardware
//
#include <hip/hip_runtime.h>
#include <hip/hip_bf16.h>

// ---------------------------------------------------------------------------
// TradeNet on MI455X (gfx1250):
//   conv1d as implicit GEMM (256 x 16321 x 16384) via V_WMMA_F32_16X16X32_F16,
//   tiles fed by the Tensor Data Mover (tensor_load_to_lds) with double
//   buffering on TENSORcnt, then an in-place sequential tanh scan.
// ---------------------------------------------------------------------------

typedef _Float16 h8_t  __attribute__((ext_vector_type(8)));
typedef _Float16 h16_t __attribute__((ext_vector_type(16)));
typedef float    f8_t  __attribute__((ext_vector_type(8)));
typedef unsigned int u32x4 __attribute__((ext_vector_type(4)));
typedef int      i32x4 __attribute__((ext_vector_type(4)));
typedef int      i32x8 __attribute__((ext_vector_type(8)));

#define NM   256      // markets / channels
#define NT   16384    // time steps
#define NW   64       // conv window
#define NL   16321    // NT - NW + 1
#define NLP1 16322    // NL + 1
#define WXN  4194304  // NM*NM*NW == NM*NT
#define XPAD 64       // zero pad past x_h (Hankel tile overreach on last t-tile)

#define TN_USE_TDM __has_builtin(__builtin_amdgcn_tensor_load_to_lds)

// ------------------------- f32 -> f16 conversion ---------------------------
__global__ void tn_cvt_f16(const float* __restrict__ src, _Float16* __restrict__ dst,
                           int n, int ntotal) {
    int i = blockIdx.x * blockDim.x + threadIdx.x;
    if (i < n)            dst[i] = (_Float16)src[i];
    else if (i < ntotal)  dst[i] = (_Float16)0.0f;
}

#if TN_USE_TDM
// Issue one TDM 2D tile load: 32 (dim0) x 64 (dim1) f16 elements into LDS,
// LDS rows padded +4 dwords per 16 dwords -> LDS row stride 40 halves (80 B).
// Descriptor layout per CDNA5 ISA.
__device__ __forceinline__ void tn_tdm_tile(unsigned lds_off, const _Float16* gptr,
                                            unsigned tdim0, unsigned tdim1,
                                            unsigned stride0) {
    unsigned long long g = (unsigned long long)(uintptr_t)gptr;
    u32x4 g0;
    g0.x = 1u;                                        // count=1 (valid), user mode
    g0.y = lds_off;                                   // LDS byte address
    g0.z = (unsigned)(g & 0xFFFFFFFFu);               // global_addr[31:0]
    g0.w = (unsigned)((g >> 32) & 0x01FFFFFFu)        // global_addr[56:32]
         | (2u << 30);                                // type=2 ("image")
    i32x8 g1;
    g1[0] = (int)((1u << 16)                          // data_size = 2 bytes
                | (1u << 20)                          // pad_enable
                | (3u << 22)                          // pad_interval: every 16 dwords
                | (3u << 25));                        // pad_amount: 4 dwords
    g1[1] = (int)((tdim0 & 0xFFFFu) << 16);           // tensor_dim0[15:0]
    g1[2] = (int)(((tdim0 >> 16) & 0xFFFFu)           // tensor_dim0[31:16]
                | ((tdim1 & 0xFFFFu) << 16));         // tensor_dim1[15:0]
    g1[3] = (int)(((tdim1 >> 16) & 0xFFFFu)           // tensor_dim1[31:16]
                | (32u << 16));                       // tile_dim0 = 32
    g1[4] = 64;                                       // tile_dim1 = 64, tile_dim2 = 0
    g1[5] = (int)stride0;                             // tensor_dim0_stride[31:0]
    g1[6] = 0;                                        // stride0 hi / stride1 lo
    g1[7] = 0;
    i32x4 z4 = {0, 0, 0, 0};
#if __clang_major__ >= 23
    i32x8 z8 = {0, 0, 0, 0, 0, 0, 0, 0};
    __builtin_amdgcn_tensor_load_to_lds(g0, g1, z4, z4, z8, 0);
#else
    __builtin_amdgcn_tensor_load_to_lds(g0, g1, z4, z4, 0);
#endif
}
#endif // TN_USE_TDM

// ------------------------- implicit-GEMM conv kernel -----------------------
// Grid: (256 t-blocks, 4 o-blocks), 256 threads (8 wave32).
// Block tile: 64(o) x 64(t); K loop over 16384 in steps of 32 (i fixed/chunk).
__global__ __launch_bounds__(256) void tn_conv_wmma(
    const _Float16* __restrict__ w_h,   // [256][16384]  (o, i*64+k)
    const _Float16* __restrict__ x_h,   // [256][16384]  (+64 zero pad)
    const float*    __restrict__ conv_b,
    float*          __restrict__ out)   // y stored at out[o*16322 + 1 + t]
{
    __shared__ __align__(16) _Float16 Abuf[2][64][40];  // [o][k], stride 40
    __shared__ __align__(16) _Float16 Bbuf[2][64][40];  // [t][k], stride 40

    const int th   = threadIdx.x;
    const int bo   = blockIdx.y << 6;
    const int t0   = blockIdx.x << 6;
    const int lane = th & 31;
    const int lo   = lane & 15;
    const int hi   = lane >> 4;
    const int wid  = th >> 5;
    const int wo   = (wid >> 1) << 4;        // 0,16,32,48
    const int wt   = (wid & 1) << 5;         // 0,32

    f8_t acc0 = {};
    f8_t acc1 = {};

#if TN_USE_TDM
    const unsigned ldsA0 = (unsigned)(uintptr_t)&Abuf[0][0][0];
    const unsigned ldsA1 = (unsigned)(uintptr_t)&Abuf[1][0][0];
    const unsigned ldsB0 = (unsigned)(uintptr_t)&Bbuf[0][0][0];
    const unsigned ldsB1 = (unsigned)(uintptr_t)&Bbuf[1][0][0];

    if (wid == 0) {
        // prologue: stage kb=0 into buffer 0
        tn_tdm_tile(ldsA0, w_h + ((size_t)bo << 14), NT, NM, NT);          // A tile
        tn_tdm_tile(ldsB0, x_h + (size_t)t0, 1u << 24, 1u << 20, 1u);     // Hankel B
    }
#else
    const int row = th >> 2;
    const int seg = th & 3;
#endif

    for (int kb = 0; kb < 512; ++kb) {
        const int cur = kb & 1;

#if TN_USE_TDM
        if (wid == 0) {
            if (kb + 1 < 512) {
                const int nb = cur ^ 1;
                const int i1 = (kb + 1) >> 1;
                const int k1 = ((kb + 1) & 1) << 5;
                tn_tdm_tile(nb ? ldsA1 : ldsA0,
                            w_h + ((size_t)bo << 14) + ((kb + 1) << 5), NT, NM, NT);
                tn_tdm_tile(nb ? ldsB1 : ldsB0,
                            x_h + ((size_t)i1 << 14) + (size_t)(k1 + t0),
                            1u << 24, 1u << 20, 1u);
                __builtin_amdgcn_s_wait_tensorcnt(2);   // current pair landed
            } else {
                __builtin_amdgcn_s_wait_tensorcnt(0);
            }
        }
#else
        {   // fallback: direct staging (A vectorized, B 2B-aligned gather)
            const int i0 = kb >> 1;
            const int k0 = (kb & 1) << 5;
            *(h8_t*)&Abuf[cur][row][seg << 3] =
                *(const h8_t*)(w_h + ((size_t)(bo + row) << 14) + (kb << 5) + (seg << 3));
            const _Float16* sp = x_h + ((size_t)i0 << 14) + (size_t)(k0 + t0 + row + (seg << 3));
            h8_t tmp;
            #pragma unroll
            for (int j = 0; j < 8; ++j) tmp[j] = sp[j];
            *(h8_t*)&Bbuf[cur][row][seg << 3] = tmp;
        }
#endif
        __syncthreads();

        // ---- fragments: contiguous k = 16*hi + s, same order in A and B ----
        union { h16_t v; h8_t h[2]; } a, b0, b1;
        a.h[0] = *(const h8_t*)&Abuf[cur][wo + lo][(hi << 4)];
        a.h[1] = *(const h8_t*)&Abuf[cur][wo + lo][(hi << 4) + 8];
        const int ko = hi << 4;
        b0.h[0] = *(const h8_t*)&Bbuf[cur][wt + lo][ko];
        b0.h[1] = *(const h8_t*)&Bbuf[cur][wt + lo][ko + 8];
        b1.h[0] = *(const h8_t*)&Bbuf[cur][wt + 16 + lo][ko];
        b1.h[1] = *(const h8_t*)&Bbuf[cur][wt + 16 + lo][ko + 8];

        acc0 = __builtin_amdgcn_wmma_f32_16x16x32_f16(
                   false, a.v, false, b0.v, (short)0, acc0, false, false);
        acc1 = __builtin_amdgcn_wmma_f32_16x16x32_f16(
                   false, a.v, false, b1.v, (short)0, acc1, false, false);

        __syncthreads();
    }

    // ---- store D (+bias): lane<16 -> col lo, M=r; lane>=16 -> col lo, M=8+r
    const int rowg = bo + wo + (hi << 3);
    #pragma unroll
    for (int r = 0; r < 8; ++r) {
        const int og   = rowg + r;
        const float bv = conv_b[og];
        const int c0 = t0 + wt + lo;
        const int c1 = t0 + wt + 16 + lo;
        if (c0 < NL) out[(size_t)og * NLP1 + 1 + c0] = acc0[r] + bv;
        if (c1 < NL) out[(size_t)og * NLP1 + 1 + c1] = acc1[r] + bv;
    }
}

// ------------------------- sequential tanh scan ----------------------------
__global__ void tn_scan(const float* __restrict__ pos_weight, float* __restrict__ out) {
    const int m = threadIdx.x;
    const float pw = pos_weight[m];
    const size_t base = (size_t)m * NLP1;
    float pos = -1.0f;
    out[base] = -1.0f;
    for (int t = 0; t < NL; ++t) {
        const float y = out[base + 1 + t];
        pos = tanhf(pos * pw + y);
        out[base + 1 + t] = pos;
    }
}

// ---------------------------------------------------------------------------
extern "C" void kernel_launch(void* const* d_in, const int* in_sizes, int n_in,
                              void* d_out, int out_size, void* d_ws, size_t ws_size,
                              hipStream_t stream) {
    const float* x  = (const float*)d_in[0];
    const float* w  = (const float*)d_in[1];
    const float* cb = (const float*)d_in[2];
    const float* pw = (const float*)d_in[3];
    float* out = (float*)d_out;

    _Float16* w_h = (_Float16*)d_ws;
    _Float16* x_h = w_h + WXN;

    tn_cvt_f16<<<(WXN + 255) / 256, 256, 0, stream>>>(w, w_h, WXN, WXN);
    tn_cvt_f16<<<(WXN + XPAD + 255) / 256, 256, 0, stream>>>(x, x_h, WXN, WXN + XPAD);

    dim3 grid(NT / 64, NM / 64);   // (256, 4)
    tn_conv_wmma<<<grid, 256, 0, stream>>>(w_h, x_h, cb, out);

    tn_scan<<<1, NM, 0, stream>>>(pw, out);
}
